// HindsightWarp_55654186221784
// MI455X (gfx1250) — matrix operand order repaired and checked by reading, not compiled, not run on hardware
//
#include <hip/hip_runtime.h>

typedef __attribute__((ext_vector_type(16))) __bf16        v16bf;
typedef __attribute__((ext_vector_type(8)))  float         v8f;
typedef __attribute__((ext_vector_type(4)))  unsigned int  u32x4;
typedef __attribute__((ext_vector_type(8)))  int           i32x8;
typedef __attribute__((ext_vector_type(4)))  int           i32x4;

#define Bn   2
#define Cn   32
#define Hn   512
#define Wn   512
#define HWn  (Hn * Wn)
#define KTOT 450
#define KPAD 480

#define TILE_X 64      // pixels per conv block (4 waves x 16)
#define TCOLS  66      // TILE_X + 2 halo columns
#define TCOLP  68      // padded LDS column stride (shorts)

// ---------- helpers ----------

__device__ __forceinline__ unsigned short f2bf_bits(float f) {
  unsigned int u = __builtin_bit_cast(unsigned int, f);
  unsigned int r = u + 0x7FFFu + ((u >> 16) & 1u);   // round-to-nearest-even
  return (unsigned short)(r >> 16);
}
__device__ __forceinline__ __bf16 bits2bf(unsigned short s) {
  return __builtin_bit_cast(__bf16, s);
}

// bilinear sample, edge-clamped (resize semantics)
__device__ __forceinline__ float bilin_cl(const float* __restrict__ p,
                                          long rowS, long xS, float x, float y) {
  float xf = floorf(x), yf = floorf(y);
  int x0 = (int)xf, y0 = (int)yf;
  float wx = x - xf, wy = y - yf;
  float acc = 0.f;
#pragma unroll
  for (int dy = 0; dy < 2; ++dy) {
#pragma unroll
    for (int dx = 0; dx < 2; ++dx) {
      int xi = min(max(x0 + dx, 0), Wn - 1);
      int yi = min(max(y0 + dy, 0), Hn - 1);
      float w = (dx ? wx : 1.f - wx) * (dy ? wy : 1.f - wy);
      acc += w * p[(long)yi * rowS + (long)xi * xS];
    }
  }
  return acc;
}

// bilinear sample, zero outside
__device__ __forceinline__ float bilin_zp(const float* __restrict__ p,
                                          long rowS, long xS, float x, float y) {
  float xf = floorf(x), yf = floorf(y);
  int x0 = (int)xf, y0 = (int)yf;
  float wx = x - xf, wy = y - yf;
  float acc = 0.f;
#pragma unroll
  for (int dy = 0; dy < 2; ++dy) {
#pragma unroll
    for (int dx = 0; dx < 2; ++dx) {
      int xi = x0 + dx, yi = y0 + dy;
      if (xi >= 0 && xi < Wn && yi >= 0 && yi < Hn) {
        float w = (dx ? wx : 1.f - wx) * (dy ? wy : 1.f - wy);
        acc += w * p[(long)yi * rowS + (long)xi * xS];
      }
    }
  }
  return acc;
}

// rollback=1 fused up->warp->down geometry: 4 hi-res subsamples, 4 bilinear
// corners each -> 16 (offset, weight) pairs; invalid corners get weight 0.
// Weights fold in the 0.25 downsample average.
__device__ __forceinline__ void make_gather16(const float* __restrict__ rx,
                                              const float* __restrict__ ry,
                                              long rowS, long xS, int y, int x,
                                              int* __restrict__ off,
                                              float* __restrict__ w) {
#pragma unroll
  for (int s = 0; s < 4; ++s) {
    int hy = 2 * y + (s >> 1), hx = 2 * x + (s & 1);
    float mx = hx * 0.5f - 0.25f, my = hy * 0.5f - 0.25f;   // hi-res -> 512
    float rxv = 2.f * bilin_cl(rx, rowS, xS, mx, my);       // upsampled res * f
    float ryv = 2.f * bilin_cl(ry, rowS, xS, mx, my);
    float smx = ((float)hx + rxv) * 0.5f - 0.25f;           // warp, map back
    float smy = ((float)hy + ryv) * 0.5f - 0.25f;
    float xf = floorf(smx), yf = floorf(smy);
    int x0 = (int)xf, y0 = (int)yf;
    float fx = smx - xf, fy = smy - yf;
#pragma unroll
    for (int dy = 0; dy < 2; ++dy) {
#pragma unroll
      for (int dx = 0; dx < 2; ++dx) {
        int j = s * 4 + dy * 2 + dx;
        int xi = x0 + dx, yi = y0 + dy;
        bool ok = (xi >= 0) & (xi < Wn) & (yi >= 0) & (yi < Hn);
        off[j] = ok ? (yi * Wn + xi) : 0;
        w[j]   = ok ? 0.25f * (dx ? fx : 1.f - fx) * (dy ? fy : 1.f - fy) : 0.f;
      }
    }
  }
}

// ---------- stage 0: pack conv weights bf16, N-major [16][KPAD] ----------

__global__ void prep_w_kernel(const float* __restrict__ Wi,
                              unsigned short* __restrict__ Wb) {
  int idx = blockIdx.x * blockDim.x + threadIdx.x;
  if (idx >= 16 * KPAD) return;
  int n = idx / KPAD, k = idx % KPAD;
  float v = 0.f;
  if (k < KTOT && n < 2) {
    int c = k / 9, tap = k % 9;                 // K = c*9 + (ty*3+tx)
    v = Wi[((long)n * 50 + c) * 9 + tap];       // W_inner[n, c, ty, tx]
  }
  Wb[idx] = f2bf_bits(v);
}

// ---------- stage 1: warped_tgt = res_warp_img(tgt, res_hw, rollback=1) ----------

__global__ void warp_img_ws_kernel(const float* __restrict__ x,
                                   const float* __restrict__ res,
                                   float* __restrict__ warped) {
  long idx = (long)blockIdx.x * blockDim.x + threadIdx.x;
  if (idx >= (long)Bn * HWn) return;
  int xx = (int)(idx % Wn);
  int yy = (int)((idx / Wn) % Hn);
  int b  = (int)(idx / HWn);

  const float* rx = res + (long)(b * 2 + 0) * HWn;
  const float* ry = res + (long)(b * 2 + 1) * HWn;

  int   off[16];
  float w[16];
  make_gather16(rx, ry, Wn, 1, yy, xx, off, w);

  long pix = (long)yy * Wn + xx;
#pragma unroll 4
  for (int c = 0; c < 16; ++c) {
    const float* img = x + (long)(b * Cn + 16 + c) * HWn;   // tgt channel
    if (c + 1 < 16)
      __builtin_prefetch(img + HWn + pix, 0, 3);
    float acc = 0.f;
#pragma unroll
    for (int j = 0; j < 16; ++j)
      acc = fmaf(w[j], img[off[j]], acc);
    warped[(long)(b * 16 + c) * HWn + pix] = acc;
  }
}

// ---------- stage 2: adj = conv3x3(bundle[50ch], W) + b  via WMMA bf16 ----------
// Block = 128 threads = 4 waves, covers a 64-pixel strip of one row.
// Bundle halo tile staged to LDS as bf16; weights DMA'd to LDS via TDM.

__global__ __launch_bounds__(128) void conv_wmma_kernel(
    const float* __restrict__ x, const float* __restrict__ warped,
    const float* __restrict__ res, const unsigned short* __restrict__ Wb,
    const float* __restrict__ bias, float* __restrict__ adj) {
  __shared__ unsigned short sW[16 * KPAD];        // 15360 B, N-major [n][k]
  __shared__ unsigned short sT[50 * 3 * TCOLP];   // 20400 B bundle tile (bf16)
  __shared__ short          sOffA[KPAD];          // per-K LDS offset (or -1)

  int tid = threadIdx.x;
  const int XT = Wn / TILE_X;                     // 8 strips per row
  int xt = blockIdx.x % XT;
  int y  = (blockIdx.x / XT) % Hn;
  int b  = blockIdx.x / (XT * Hn);
  int X0 = xt * TILE_X;

  // --- wave 0: TDM copy of packed weights (15360 B, 1-D tile) into LDS ---
  if (tid < 32) {
    unsigned lds_addr = (unsigned)(uintptr_t)(&sW[0]);
    unsigned long long ga = (unsigned long long)(uintptr_t)Wb;
    // D# group 0: count=1 | lds_addr | global_addr | type=2
    u32x4 g0 = { 1u,
                 lds_addr,
                 (unsigned)(ga & 0xFFFFFFFFu),
                 (unsigned)((ga >> 32) & 0x1FFFFFFu) | (2u << 30) };
    // D# group 1: data_size=1(2B); tensor_dim0=tile_dim0=7680; dim1=1; stride=7680
    i32x8 g1 = { 0x00010000,      // mask=0, data_size=1 (2 bytes)
                 0x1E000000,      // tensor_dim0[15:0]=0x1E00 @ [31:16]
                 0x00010000,      // tensor_dim0[31:16]=0; tensor_dim1[15:0]=1
                 0x1E000000,      // tensor_dim1 hi=0; tile_dim0=0x1E00 @ [31:16]
                 0x00000001,      // tile_dim1=1; tile_dim2=0
                 7680,            // tensor_dim0_stride[31:0]
                 0, 0 };
    i32x4 z4 = { 0, 0, 0, 0 };
    i32x8 z8 = { 0, 0, 0, 0, 0, 0, 0, 0 };
    // 6-arg form on this toolchain: (g0, g1, g2, g3, g?, cpol)
    __builtin_amdgcn_tensor_load_to_lds(g0, g1, z4, z4, z8, 0);
    __builtin_amdgcn_s_wait_tensorcnt(0);
  }

  // --- per-K A-fragment LDS offset table (kills div/mod in the hot loop) ---
  for (int k = tid; k < KPAD; k += 128) {
    short o = -1;
    if (k < KTOT) {
      int c = k / 9, tap = k % 9;
      o = (short)((c * 3 + tap / 3) * TCOLP + (tap % 3));   // add px_local at use
    }
    sOffA[k] = o;
  }

  // --- cooperative stage of 50ch x 3row x 66col halo tile, f32 -> bf16 LDS ---
  for (int e = tid; e < 50 * 3 * TCOLS; e += 128) {
    int col = e % TCOLS;
    int t2  = e / TCOLS;
    int row = t2 % 3;
    int c   = t2 / 3;
    int gx = X0 + col - 1;
    int gy = y + row - 1;
    float v = 0.f;
    if (gx >= 0 && gx < Wn && gy >= 0 && gy < Hn) {
      const float* p;
      if (c < 32)      p = x      + (long)(b * Cn + c) * HWn;        // src|tgt
      else if (c < 48) p = warped + (long)(b * 16 + (c - 32)) * HWn; // warped_tgt
      else             p = res    + (long)(b * 2  + (c - 48)) * HWn; // res
      v = p[(long)gy * Wn + gx];
    }
    sT[(c * 3 + row) * TCOLP + col] = f2bf_bits(v);
  }
  __syncthreads();

  // --- WMMA: one wave per 16-pixel x 16-outchan tile ---
  int wv = tid >> 5;
  int t  = tid & 31;
  int hi = t >> 4;           // lane half per ISA A/B layout
  int m  = t & 15;           // A row (pixel); also B/D column N
  int px_local = wv * 16 + m + 1;   // LDS column of this lane's center pixel

  v8f acc = {};
  for (int kb = 0; kb < KPAD; kb += 32) {
    v16bf a;
#pragma unroll
    for (int i = 0; i < 16; ++i) {
      // A (16-bit, 16x32): half-slot i -> K = i (+16 lane>=16), matching B
      int kl = i + (hi ? 16 : 0);
      short o = sOffA[kb + kl];
      unsigned short bits = (o >= 0) ? sT[o + px_local - 1] : (unsigned short)0;
      a[i] = bits2bf(bits);
    }
    // B (16-bit, 32x16): half-slot i -> K = i (+16 lane>=16), lane&15 -> N.
    // N-major LDS layout makes the fragment one contiguous 32B chunk.
    v16bf bm = *(const v16bf*)&sW[m * KPAD + kb + (hi ? 16 : 0)];
    acc = __builtin_amdgcn_wmma_f32_16x16x32_bf16(false, a, false, bm,
                                                  (short)0, acc, false, false);
  }

  // D: lane holds column N = t&15, rows M = r + 8*hi
  if (m < 2) {
    float bv = bias[m];
#pragma unroll
    for (int r = 0; r < 8; ++r) {
      int mm = r + (hi ? 8 : 0);
      adj[((long)(b * 2 + m) * Hn + y) * Wn + (X0 + wv * 16 + mm)] = acc[r] + bv;
    }
  }
}

// ---------- stage 3: new_res = adj_hw + res_warp_res(res_hw, adj_hw, 1) ----------

__global__ void combine_kernel(const float* __restrict__ res,
                               const float* __restrict__ adj,
                               float* __restrict__ newres) {
  long idx = (long)blockIdx.x * blockDim.x + threadIdx.x;
  if (idx >= (long)Bn * HWn) return;
  int xx = (int)(idx % Wn);
  int yy = (int)((idx / Wn) % Hn);
  int b  = (int)(idx / HWn);

  const float* ax = adj + (long)(b * 2 + 0) * HWn;
  const float* ay = adj + (long)(b * 2 + 1) * HWn;
  const float* rx = res + (long)(b * 2 + 0) * HWn;
  const float* ry = res + (long)(b * 2 + 1) * HWn;

  float accx = 0.f, accy = 0.f;
#pragma unroll
  for (int s = 0; s < 4; ++s) {
    int hy = 2 * yy + (s >> 1), hx = 2 * xx + (s & 1);
    float mx = hx * 0.5f - 0.25f, my = hy * 0.5f - 0.25f;
    float bx = 2.f * bilin_cl(ax, Wn, 1, mx, my);   // upsampled adj * 2
    float by = 2.f * bilin_cl(ay, Wn, 1, mx, my);
    float smx = ((float)hx + bx) * 0.5f - 0.25f;
    float smy = ((float)hy + by) * 0.5f - 0.25f;
    accx += 2.f * bilin_zp(rx, Wn, 1, smx, smy);    // sample upsampled res (*2)
    accy += 2.f * bilin_zp(ry, Wn, 1, smx, smy);
  }
  long o = ((long)b * HWn + (long)yy * Wn + xx) * 2;
  newres[o + 0] = ax[(long)yy * Wn + xx] + 0.125f * accx;  // 0.25 avg * 0.5 scale
  newres[o + 1] = ay[(long)yy * Wn + xx] + 0.125f * accy;
}

// ---------- stage 4: out = concat(src, res_warp_img(tgt, new_res, 1)) ----------

__global__ void final_warp_kernel(const float* __restrict__ x,
                                  const float* __restrict__ newres,
                                  float* __restrict__ out) {
  long idx = (long)blockIdx.x * blockDim.x + threadIdx.x;
  if (idx >= (long)Bn * HWn) return;
  int xx = (int)(idx % Wn);
  int yy = (int)((idx / Wn) % Hn);
  int b  = (int)(idx / HWn);

  const float* rx = newres + (long)b * HWn * 2 + 0;   // interleaved HWC
  const float* ry = newres + (long)b * HWn * 2 + 1;

  int   off[16];
  float w[16];
  make_gather16(rx, ry, (long)Wn * 2, 2, yy, xx, off, w);

  long pix = (long)yy * Wn + xx;
#pragma unroll 4
  for (int c = 0; c < 16; ++c) {
    out[(long)(b * Cn + c) * HWn + pix] = x[(long)(b * Cn + c) * HWn + pix]; // src
    const float* img = x + (long)(b * Cn + 16 + c) * HWn;
    if (c + 1 < 16)
      __builtin_prefetch(img + HWn + pix, 0, 3);
    float acc = 0.f;
#pragma unroll
    for (int j = 0; j < 16; ++j)
      acc = fmaf(w[j], img[off[j]], acc);
    out[(long)(b * Cn + 16 + c) * HWn + pix] = acc;
  }
}

// ---------- launcher ----------

extern "C" void kernel_launch(void* const* d_in, const int* in_sizes, int n_in,
                              void* d_out, int out_size, void* d_ws, size_t ws_size,
                              hipStream_t stream) {
  const float* x   = (const float*)d_in[0];   // [2,32,512,512]
  const float* res = (const float*)d_in[1];   // [2,2,512,512]
  const float* Wi  = (const float*)d_in[2];   // [2,50,3,3]
  const float* bi  = (const float*)d_in[3];   // [2]
  float* out = (float*)d_out;

  char* ws = (char*)d_ws;
  unsigned short* Wb = (unsigned short*)ws;                       // 15.4 KB
  float* warped = (float*)(ws + (1 << 16));                       // 32 MB
  float* adj    = (float*)((char*)warped + (size_t)Bn * 16 * HWn * sizeof(float));
  float* newres = (float*)((char*)adj    + (size_t)Bn * 2  * HWn * sizeof(float));

  prep_w_kernel<<<(16 * KPAD + 255) / 256, 256, 0, stream>>>(Wi, Wb);
  warp_img_ws_kernel<<<(Bn * HWn) / 256, 256, 0, stream>>>(x, res, warped);
  conv_wmma_kernel<<<Bn * Hn * (Wn / TILE_X), 128, 0, stream>>>(
      x, warped, res, Wb, bi, adj);
  combine_kernel<<<(Bn * HWn) / 256, 256, 0, stream>>>(res, adj, newres);
  final_warp_kernel<<<(Bn * HWn) / 256, 256, 0, stream>>>(x, newres, out);
}
